// DGCNN_90503550861614
// MI455X (gfx1250) — matrix operand order repaired and checked by reading, not compiled, not run on hardware
//
#include <hip/hip_runtime.h>
#include <hip/hip_bf16.h>
#include <math.h>

typedef __attribute__((ext_vector_type(16))) _Float16 v16h;
typedef __attribute__((ext_vector_type(8)))  _Float16 v8h;
typedef __attribute__((ext_vector_type(8)))  float    v8f;

#define NNODES   131072        // 256 * 512
#define NGRAPH   256
#define NPER     512
#define DIMF     128
#define FEATD    97
#define KPOOL    64

// ---------------------------------------------------------------------------
// WMMA GEMM: H[m,n] = act( sum_k A[m,k] * W[n,k] + bias[n] )
//   A: Mrows x Cin, row stride lda, f32 (AF32=true) or f16 (AF32=false)
//   W: f16, zero-padded to ceil(Cout/16)*16 rows x Cin  (branch-free B loads)
// Per ISA 7.12.2 f16 layouts:
//   A lane: two contiguous K-runs of 8  ([half*8, +8) and [16+half*8, +8))
//   B lane: same K-runs as A ([half*8, +8) and [16+half*8, +8)), col = lane&15
//   D: M = r + half*8, N = lane&15
// One wave per 16x16 tile, 8 waves per block. All WMMA branches wave-uniform.
// ---------------------------------------------------------------------------
template <bool AF32>
__global__ void wmma_gemm_kernel(const void* __restrict__ Aq, int lda,
                                 const _Float16* __restrict__ W,
                                 const float* __restrict__ bias,
                                 float* __restrict__ H, int ldh,
                                 float* __restrict__ ACC,
                                 int Mrows, int Cin, int Cout, int act) {
    const int lane = threadIdx.x & 31;
    const int wave = threadIdx.x >> 5;
    const int m0   = (blockIdx.x * (blockDim.x >> 5) + wave) * 16;
    const int n0   = blockIdx.y * 16;
    if (m0 >= Mrows) return;                 // wave-uniform; EXEC stays full

    const int row  = lane & 15;
    const int half = lane >> 4;

    const _Float16* brow = W + (size_t)(n0 + row) * Cin;   // padded: always valid
    v8f c = {0.f, 0.f, 0.f, 0.f, 0.f, 0.f, 0.f, 0.f};

    for (int k0 = 0; k0 < Cin; k0 += 32) {
        v16h a, b;
        {
            v8h blo = *(const v8h*)(brow + k0 + half * 8);
            v8h bhi = *(const v8h*)(brow + k0 + 16 + half * 8);
            b = __builtin_shufflevector(blo, bhi, 0, 1, 2, 3, 4, 5, 6, 7,
                                        8, 9, 10, 11, 12, 13, 14, 15);
        }
        if (AF32) {
            const float* ap = (const float*)Aq + (size_t)(m0 + row) * lda
                              + k0 + half * 8;
            const float4 l0 = *(const float4*)(ap);
            const float4 l1 = *(const float4*)(ap + 4);
            const float4 h0 = *(const float4*)(ap + 16);
            const float4 h1 = *(const float4*)(ap + 20);
            a[0]  = (_Float16)l0.x; a[1]  = (_Float16)l0.y;
            a[2]  = (_Float16)l0.z; a[3]  = (_Float16)l0.w;
            a[4]  = (_Float16)l1.x; a[5]  = (_Float16)l1.y;
            a[6]  = (_Float16)l1.z; a[7]  = (_Float16)l1.w;
            a[8]  = (_Float16)h0.x; a[9]  = (_Float16)h0.y;
            a[10] = (_Float16)h0.z; a[11] = (_Float16)h0.w;
            a[12] = (_Float16)h1.x; a[13] = (_Float16)h1.y;
            a[14] = (_Float16)h1.z; a[15] = (_Float16)h1.w;
        } else {
            const _Float16* ap = (const _Float16*)Aq + (size_t)(m0 + row) * lda
                                 + k0 + half * 8;
            v8h alo = *(const v8h*)(ap);
            v8h ahi = *(const v8h*)(ap + 16);
            a = __builtin_shufflevector(alo, ahi, 0, 1, 2, 3, 4, 5, 6, 7,
                                        8, 9, 10, 11, 12, 13, 14, 15);
        }
        c = __builtin_amdgcn_wmma_f32_16x16x32_f16(
                /*neg_a=*/false, a, /*neg_b=*/false, b,
                /*c_mod=*/(short)0, c, /*reuse_a=*/false, /*reuse_b=*/false);
    }

    const int n = n0 + row;
    if (n < Cout) {
        const float bv = bias[n];
#pragma unroll
        for (int r = 0; r < 8; ++r) {
            const int m = m0 + r + half * 8;
            float val = c[r] + bv;
            if (act == 1) val = fmaxf(val, 0.f);
            H[(size_t)m * ldh + n] = val;
            if (ACC) ACC[(size_t)m * ldh + n] = val;
        }
    }
}

// ---- f32 -> f16 convert with zero padding (weights / small tensors) --------
__global__ void cvt_pad_f16_kernel(const float* __restrict__ src,
                                   _Float16* __restrict__ dst,
                                   int srcRows, int srcCols,
                                   int dstRows, int dstCols) {
    int i = blockIdx.x * blockDim.x + threadIdx.x;
    if (i >= dstRows * dstCols) return;
    int r = i / dstCols, cc = i - r * dstCols;
    float v = (r < srcRows && cc < srcCols) ? src[(size_t)r * srcCols + cc] : 0.f;
    dst[i] = (_Float16)v;
}

// ---- degree: deg[i] = 1 (self loop) + #edges with src == i -----------------
__global__ void deg_init_kernel(float* __restrict__ deg, int n) {
    int i = blockIdx.x * blockDim.x + threadIdx.x;
    if (i < n) deg[i] = 1.0f;
}
__global__ void deg_count_kernel(float* __restrict__ deg,
                                 const int* __restrict__ src, int E) {
    int e = blockIdx.x * blockDim.x + threadIdx.x;
    if (e < E) atomicAdd(&deg[src[e]], 1.0f);
}

// ---- edge scatter: acc[dst] += h[src]  (acc pre-seeded with h) -------------
__global__ void scatter_add_kernel(const float* __restrict__ h,
                                   float* __restrict__ acc,
                                   const int* __restrict__ src,
                                   const int* __restrict__ dst,
                                   int E, int C) {
    int e = blockIdx.x * blockDim.x + threadIdx.x;
    if (e >= E) return;
    const float* hs = h   + (size_t)src[e] * C;
    float*       ad = acc + (size_t)dst[e] * C;
    for (int c = 0; c < C; ++c) atomicAdd(&ad[c], hs[c]);
}

// ---- v = tanh(acc/deg); write f32 feat slice + dense f16 copy --------------
__global__ void finalize_kernel(const float* __restrict__ acc,
                                const float* __restrict__ deg,
                                float* __restrict__ featOut, int ldf,
                                _Float16* __restrict__ h16,
                                int C, int n) {
    int i = blockIdx.x * blockDim.x + threadIdx.x;
    int node = i / C, c = i - node * C;
    if (node >= n) return;
    float v = tanhf(acc[(size_t)node * C + c] / deg[node]);
    featOut[(size_t)node * ldf + c] = v;
    if (h16) h16[(size_t)node * C + c] = (_Float16)v;
}

// ---- per-graph sort pool: bitonic sort 512 keys desc; emit padded f16 ------
__global__ void sortpool_kernel(const float* __restrict__ feat,
                                _Float16* __restrict__ topk16) {
    __shared__ float key[NPER];
    __shared__ int   idx[NPER];
    const int b = blockIdx.x;
    const int t = threadIdx.x;
    const float* gf = feat + (size_t)b * NPER * FEATD;
    key[t] = gf[(size_t)t * FEATD + (FEATD - 1)];
    idx[t] = t;
    __syncthreads();
    for (int k = 2; k <= NPER; k <<= 1) {
        for (int j = k >> 1; j > 0; j >>= 1) {
            int ixj = t ^ j;
            if (ixj > t) {
                bool desc = ((t & k) == 0);
                bool sw = desc ? (key[t] < key[ixj]) : (key[t] > key[ixj]);
                if (sw) {
                    float tk = key[t]; key[t] = key[ixj]; key[ixj] = tk;
                    int   ti = idx[t]; idx[t] = idx[ixj]; idx[ixj] = ti;
                }
            }
            __syncthreads();
        }
    }
    // gather top-64 rows, zero-pad 97 -> 128 cols, f16 for WMMA conv1
    for (int p = t; p < KPOOL * 128; p += NPER) {
        int kk = p >> 7, d = p & 127;
        float v = (d < FEATD) ? gf[(size_t)idx[kk] * FEATD + d] : 0.f;
        topk16[((size_t)b * KPOOL + kk) * 128 + d] = (_Float16)v;
    }
}

// ---- relu + maxpool(2) over conv1 output [16384,16] -> [256,16,32] ---------
__global__ void pool_kernel(const float* __restrict__ c1,
                            float* __restrict__ pool) {
    int i = blockIdx.x * blockDim.x + threadIdx.x;
    if (i >= NGRAPH * 16 * 32) return;
    const int kk = i & 31;
    const int o  = (i >> 5) & 15;
    const int b  = i >> 9;
    size_t base = ((size_t)b * KPOOL + kk * 2) * 16 + o;
    float v = fmaxf(c1[base], c1[base + 16]);
    pool[i] = fmaxf(v, 0.f);   // relu-then-max == max-then-relu
}

// ---- conv2: [256,16,32] -> [256,32,28], k=5, relu; emit f16 ----------------
__global__ void conv2_kernel(const float* __restrict__ pool,
                             const float* __restrict__ w,   // [32,16,5]
                             const float* __restrict__ bias,
                             _Float16* __restrict__ out) {  // [256,896] f16
    int i = blockIdx.x * blockDim.x + threadIdx.x;
    if (i >= NGRAPH * 32 * 28) return;
    const int tp = i % 28;
    const int o  = (i / 28) & 31;
    const int b  = i / (32 * 28);
    float s = bias[o];
    for (int ci = 0; ci < 16; ++ci) {
        const float* pr = pool + ((size_t)b * 16 + ci) * 32 + tp;
        const float* wr = w + ((size_t)o * 16 + ci) * 5;
        for (int kk = 0; kk < 5; ++kk) s += pr[kk] * wr[kk];
    }
    out[i] = (_Float16)fmaxf(s, 0.f);
}

// ---- dense2: [256,128] x [10,128]^T + b -> d_out [256,10] ------------------
__global__ void dense2_kernel(const float* __restrict__ hid,
                              const float* __restrict__ w,  // [10,128]
                              const float* __restrict__ bias,
                              float* __restrict__ out) {
    int i = blockIdx.x * blockDim.x + threadIdx.x;
    if (i >= NGRAPH * 10) return;
    const int j = i % 10, b = i / 10;
    const float* hr = hid + (size_t)b * 128;
    const float* wr = w + (size_t)j * 128;
    float s = bias[j];
    for (int m = 0; m < 128; ++m) s += hr[m] * wr[m];
    out[i] = s;
}

extern "C" void kernel_launch(void* const* d_in, const int* in_sizes, int n_in,
                              void* d_out, int out_size, void* d_ws, size_t ws_size,
                              hipStream_t stream) {
    const float* x    = (const float*)d_in[0];
    const float* W0   = (const float*)d_in[1];
    const float* b0   = (const float*)d_in[2];
    const float* W1   = (const float*)d_in[3];
    const float* b1   = (const float*)d_in[4];
    const float* W2   = (const float*)d_in[5];
    const float* b2   = (const float*)d_in[6];
    const float* W3   = (const float*)d_in[7];
    const float* b3   = (const float*)d_in[8];
    const float* c1w  = (const float*)d_in[9];   // [16,1,97]
    const float* c1b  = (const float*)d_in[10];
    const float* c2w  = (const float*)d_in[11];  // [32,16,5]
    const float* c2b  = (const float*)d_in[12];
    const float* d1w  = (const float*)d_in[13];  // [128,896]
    const float* d1b  = (const float*)d_in[14];
    const float* d2w  = (const float*)d_in[15];  // [10,128]
    const float* d2b  = (const float*)d_in[16];
    const int*   esrc = (const int*)d_in[17];
    const int*   edst = (const int*)d_in[18];
    const int    E    = in_sizes[17];
    float*       out  = (float*)d_out;

    // ---- workspace layout (f32 region then f16 region; all 32B aligned) ----
    float* fp = (float*)d_ws;
    float* deg   = fp; fp += NNODES;                    // [N]
    float* h     = fp; fp += (size_t)NNODES * 32;       // [N,32] lin out (f32)
    float* acc   = fp; fp += (size_t)NNODES * 32;       // [N,32] scatter acc
    float* feat  = fp; fp += (size_t)NNODES * FEATD;    // [N,97]
    float* c1out = fp; fp += (size_t)NGRAPH * KPOOL * 16; // [16384,16]
    float* pool  = fp; fp += (size_t)NGRAPH * 16 * 32;  // [256,16,32]
    float* hid   = fp; fp += (size_t)NGRAPH * 128;      // [256,128]
    _Float16* hp = (_Float16*)fp;
    _Float16* h16    = hp; hp += (size_t)NNODES * 32;        // layer out f16
    _Float16* topk16 = hp; hp += (size_t)NGRAPH * KPOOL * 128;
    _Float16* c2o16  = hp; hp += (size_t)NGRAPH * 896;
    _Float16* w0f    = hp; hp += 32 * 128;
    _Float16* w1f    = hp; hp += 32 * 32;
    _Float16* w2f    = hp; hp += 32 * 32;
    _Float16* w3f    = hp; hp += 16 * 32;                    // padded 1->16
    _Float16* c1wf   = hp; hp += 16 * 128;                   // padded 97->128
    _Float16* d1wf   = hp; hp += 128 * 896;

    const int TB = 256;
    auto blks = [](int n, int t) { return (n + t - 1) / t; };

    // weight conversions (f16, zero-padded)
    cvt_pad_f16_kernel<<<blks(32 * 128, TB), TB, 0, stream>>>(W0, w0f, 32, 128, 32, 128);
    cvt_pad_f16_kernel<<<blks(32 * 32, TB), TB, 0, stream>>>(W1, w1f, 32, 32, 32, 32);
    cvt_pad_f16_kernel<<<blks(32 * 32, TB), TB, 0, stream>>>(W2, w2f, 32, 32, 32, 32);
    cvt_pad_f16_kernel<<<blks(16 * 32, TB), TB, 0, stream>>>(W3, w3f, 1, 32, 16, 32);
    cvt_pad_f16_kernel<<<blks(16 * 128, TB), TB, 0, stream>>>(c1w, c1wf, 16, 97, 16, 128);
    cvt_pad_f16_kernel<<<blks(128 * 896, TB), TB, 0, stream>>>(d1w, d1wf, 128, 896, 128, 896);

    // degrees (edges fixed across layers)
    deg_init_kernel<<<blks(NNODES, TB), TB, 0, stream>>>(deg, NNODES);
    deg_count_kernel<<<blks(E, TB), TB, 0, stream>>>(deg, esrc, E);

    // ---- 4 GCN layers: gemm -> scatter -> tanh(acc/deg) --------------------
    // layer 0: f32 A (x), f16 W
    {
        dim3 g(NNODES / 128, 2);
        wmma_gemm_kernel<true><<<g, TB, 0, stream>>>(x, DIMF, w0f, b0,
                                                     h, 32, acc,
                                                     NNODES, DIMF, 32, 0);
        scatter_add_kernel<<<blks(E, TB), TB, 0, stream>>>(h, acc, esrc, edst, E, 32);
        finalize_kernel<<<blks(NNODES * 32, TB), TB, 0, stream>>>(
            acc, deg, feat + 0, FEATD, h16, 32, NNODES);
    }
    // layers 1..2: f16 A (h16)
    const _Float16* wmid[2] = { w1f, w2f };
    const float*    bmid[2] = { b1, b2 };
    for (int l = 0; l < 2; ++l) {
        dim3 g(NNODES / 128, 2);
        wmma_gemm_kernel<false><<<g, TB, 0, stream>>>(h16, 32, wmid[l], bmid[l],
                                                      h, 32, acc,
                                                      NNODES, 32, 32, 0);
        scatter_add_kernel<<<blks(E, TB), TB, 0, stream>>>(h, acc, esrc, edst, E, 32);
        finalize_kernel<<<blks(NNODES * 32, TB), TB, 0, stream>>>(
            acc, deg, feat + 32 + l * 32, FEATD, h16, 32, NNODES);
    }
    // layer 3: Cout=1 (padded weights)
    {
        dim3 g(NNODES / 128, 1);
        wmma_gemm_kernel<false><<<g, TB, 0, stream>>>(h16, 32, w3f, b3,
                                                      h, 1, acc,
                                                      NNODES, 32, 1, 0);
        scatter_add_kernel<<<blks(E, TB), TB, 0, stream>>>(h, acc, esrc, edst, E, 1);
        finalize_kernel<<<blks(NNODES, TB), TB, 0, stream>>>(
            acc, deg, feat + 96, FEATD, (_Float16*)nullptr, 1, NNODES);
    }

    // sort pooling -> padded f16 [16384,128]
    sortpool_kernel<<<NGRAPH, NPER, 0, stream>>>(feat, topk16);

    // conv1 as WMMA GEMM: [16384,128(pad97)] x [16,128]^T
    {
        dim3 g(NGRAPH * KPOOL / 128, 1);
        wmma_gemm_kernel<false><<<g, TB, 0, stream>>>(topk16, 128, c1wf, c1b,
                                                      c1out, 16, nullptr,
                                                      NGRAPH * KPOOL, 128, 16, 0);
    }
    pool_kernel<<<blks(NGRAPH * 16 * 32, TB), TB, 0, stream>>>(c1out, pool);

    // conv2 -> f16 flat [256,896]
    conv2_kernel<<<blks(NGRAPH * 32 * 28, TB), TB, 0, stream>>>(pool, c2w, c2b, c2o16);

    // dense1 via WMMA: [256,896] x [128,896]^T, relu
    {
        dim3 g(NGRAPH / 128, 128 / 16);
        wmma_gemm_kernel<false><<<g, TB, 0, stream>>>(c2o16, 896, d1wf, d1b,
                                                      hid, 128, nullptr,
                                                      NGRAPH, 896, 128, 1);
    }

    // dense2 -> output
    dense2_kernel<<<blks(NGRAPH * 10, TB), TB, 0, stream>>>(hid, d2w, d2b, out);
}
